// RWKVAttention_17506286698980
// MI455X (gfx1250) — matrix operand order repaired and checked by reading, not compiled, not run on hardware
//
#include <hip/hip_runtime.h>
#include <hip/hip_bf16.h>

// ---------------- problem constants ----------------
#define HH   2048
#define TT   4096
#define BB   4
#define NN   (BB * TT)        // 16384 tokens
#define LM   32               // LORA_MIX
#define LDY  64               // LORA_DECAY

typedef __attribute__((ext_vector_type(16))) _Float16 v16h;
typedef __attribute__((ext_vector_type(8)))  float    v8f;
typedef __attribute__((ext_vector_type(4)))  int      v4i;

// ---------------- tiling ----------------
#define TM    128   // token rows per block
#define TN    128   // output cols per block
#define KB    32    // K depth per step (== WMMA K depth)
#define LDS_S 40    // LDS row stride in halfs (80B rows keep 16B alignment)

// ---------------- CDNA5 async memory->LDS ----------------
#if defined(__AMDGCN__) && __has_builtin(__builtin_amdgcn_global_load_async_to_lds_b128)
#define HAS_ASYNC 1
#else
#define HAS_ASYNC 0
#endif

typedef __attribute__((address_space(1))) v4i* g128p;  // global 16B chunk
typedef __attribute__((address_space(3))) v4i* l128p;  // LDS 16B chunk

// 16B global -> LDS copy (async on CDNA5, tracked with ASYNCcnt)
__device__ __forceinline__ void async_ld16(const _Float16* g, _Float16* l) {
#if HAS_ASYNC
  __builtin_amdgcn_global_load_async_to_lds_b128(
      (g128p)(size_t)(const void*)g, (l128p)l, 0, 0);
#else
  *reinterpret_cast<uint4*>(l) = *reinterpret_cast<const uint4*>(g);
#endif
}

template <int N>
__device__ __forceinline__ void wait_async() {
#if HAS_ASYNC
#if __has_builtin(__builtin_amdgcn_s_wait_asynccnt)
  __builtin_amdgcn_s_wait_asynccnt(N);
#else
  asm volatile("s_wait_asynccnt %0" ::"i"(N) : "memory");
#endif
#endif
}

// ================= elementwise helpers =================

// LDS-tiled transpose + convert: src is K x M (f32, row-major),
// dst is M x K (f16, row-major).  B operands of every GEMM are weights,
// so transposing once makes both LDS staging and WMMA B-fragment reads
// fully vectorized (ds_load_b128) with zero scalar gathers.
__global__ void cvt_transpose(const float* __restrict__ src,
                              _Float16* __restrict__ dst, int K, int M) {
  __shared__ float tile[32][33];
  int k0 = blockIdx.y * 32, m0 = blockIdx.x * 32;
  for (int i = threadIdx.y; i < 32; i += 8) {
    int k = k0 + i, m = m0 + threadIdx.x;
    tile[i][threadIdx.x] = (k < K && m < M) ? src[(size_t)k * M + m] : 0.f;
  }
  __syncthreads();
  for (int i = threadIdx.y; i < 32; i += 8) {
    int m = m0 + i, k = k0 + threadIdx.x;
    if (m < M && k < K) dst[(size_t)m * K + k] = (_Float16)tile[threadIdx.x][i];
  }
}

// Amix[n,h] = x + (shifted - x) * time_maa_x[h], stored f16
__global__ void prep_mix(const float* __restrict__ hidden,
                         const float* __restrict__ attn_x,
                         const float* __restrict__ maa_x,
                         _Float16* __restrict__ Amix, size_t total) {
  for (size_t i = (size_t)blockIdx.x * blockDim.x + threadIdx.x; i < total;
       i += (size_t)gridDim.x * blockDim.x) {
    int h = (int)(i % HH);
    size_t n = i / HH;
    int t = (int)(n % TT);
    int b = (int)(n / TT);
    float x  = hidden[i];
    float sh = t ? hidden[i - HH] : attn_x[(size_t)b * HH + h];
    Amix[i] = (_Float16)(x + (sh - x) * maa_x[h]);
  }
}

// attn_x_new = hidden[:, T-1, :]
__global__ void copy_last(const float* __restrict__ hidden,
                          float* __restrict__ out) {
  int i = blockIdx.x * blockDim.x + threadIdx.x;   // < B*H
  if (i < BB * HH) {
    int b = i / HH, h = i % HH;
    out[i] = hidden[((size_t)b * TT + (TT - 1)) * HH + h];
  }
}

// ================= generic WMMA f16 GEMM =================
// C[n,m] = epilogue( sum_k A[n,k] * Bt[m,k] )       (Bt = W^T, f16)
// EPI: 0 = f32 store, 1 = f16 tanh, 2 = f32 silu,
//      3 = f16 (v + maa_const[m] + shifted(n,m))    [M must be H]
//      4 = f32 (v + bias[m])
// Double-buffered LDS tiles fed by GLOBAL_LOAD_ASYNC_TO_LDS_B128 (ASYNCcnt).
template <int EPI>
__global__ __launch_bounds__(256)
void gemm_wmma_f16(const _Float16* __restrict__ A, int lda,
                   const _Float16* __restrict__ Bt, int ldbt,
                   int N, int K, int M,
                   float* __restrict__ outF, _Float16* __restrict__ outH,
                   const float* __restrict__ bias,
                   const float* __restrict__ hidden,
                   const float* __restrict__ attn_x,
                   const float* __restrict__ maa_const) {
  __shared__ _Float16 lsA[2][TM * LDS_S];   // 2 x 10.0 KB
  __shared__ _Float16 lsB[2][TN * LDS_S];   // 2 x 10.0 KB

  const int tid  = threadIdx.x;
  const int lane = tid & 31;
  const int wv   = tid >> 5;    // 8 waves
  const int wm   = wv & 1;      // 2 row groups of 64
  const int wn   = wv >> 1;     // 4 col groups of 32

  const int rowBase = blockIdx.y * TM;
  const int colBase = blockIdx.x * TN;

  const int rc = tid >> 2;           // 0..63 (row within half-tile)
  const int kc = (tid & 3) << 3;     // 8-half chunk within KB

  // Issue one KB-deep tile (A: TM x KB, Bt: TN x KB) into buffer s.
  // 4 async ops per thread; OOB output-columns are clamped to row M-1
  // (branchless; those columns are never stored by the epilogue).
  auto issue = [&](int k0, int s) {
#pragma unroll
    for (int i = 0; i < 2; ++i) {
      int r = rc + i * 64;
      async_ld16(A + (size_t)(rowBase + r) * lda + (k0 + kc),
                 &lsA[s][r * LDS_S + kc]);
      int gm = colBase + r;
      if (gm >= M) gm = M - 1;
      async_ld16(Bt + (size_t)gm * ldbt + (k0 + kc),
                 &lsB[s][r * LDS_S + kc]);
    }
  };

  v8f acc[4][2] = {};   // wave tile: 64 rows x 32 cols

  const int nk = K / KB;             // K is always a multiple of KB
  issue(0, 0);
  for (int kk = 0; kk < nk; ++kk) {
    const int s = kk & 1;
    if (kk + 1 < nk) {
      issue((kk + 1) * KB, s ^ 1);   // stream next tile into other buffer
      wait_async<4>();               // older 4 ops (tile kk) complete, in order
    } else {
      wait_async<0>();
    }
    __syncthreads();

    // ---- B fragments (2 per wave), contiguous K -> ds_load_b128 x2 ----
    v16h bf[2];
#pragma unroll
    for (int tj = 0; tj < 2; ++tj) {
      const int mcol = wn * 32 + tj * 16 + (lane & 15);
      const int kb0  = (lane >> 4) << 4;   // lanes 0-15: K 0..15, 16-31: K 16..31
#pragma unroll
      for (int e = 0; e < 16; ++e)
        bf[tj][e] = lsB[s][mcol * LDS_S + kb0 + e];
    }

    // ---- 4x2 WMMA per wave ----
#pragma unroll
    for (int ti = 0; ti < 4; ++ti) {
      v16h af;
      const int nrow  = wm * 64 + ti * 16 + (lane & 15);
      const int khalf = (lane >> 4) << 4;  // lanes 0-15: K 0..15, 16-31: K 16..31
#pragma unroll
      for (int e = 0; e < 16; ++e) {
        int k = khalf + e;
        af[e] = lsA[s][nrow * LDS_S + k];
      }
#pragma unroll
      for (int tj = 0; tj < 2; ++tj)
        acc[ti][tj] = __builtin_amdgcn_wmma_f32_16x16x32_f16(
            false, af, false, bf[tj], (short)0, acc[ti][tj], false, false);
    }
    __syncthreads();
  }

  // ---- epilogue / store (f32 C layout: lane -> col, e(+8*hi) -> row) ----
#pragma unroll
  for (int ti = 0; ti < 4; ++ti)
#pragma unroll
    for (int tj = 0; tj < 2; ++tj) {
      const int ncol = colBase + wn * 32 + tj * 16 + (lane & 15);
      if (ncol >= M) continue;
#pragma unroll
      for (int e = 0; e < 8; ++e) {
        int nrow = rowBase + wm * 64 + ti * 16 + e + ((lane >> 4) << 3);
        float v = acc[ti][tj][e];
        size_t oi = (size_t)nrow * M + ncol;
        if (EPI == 0) {
          outF[oi] = v;
        } else if (EPI == 1) {
          outH[oi] = (_Float16)tanhf(v);
        } else if (EPI == 2) {
          outF[oi] = v / (1.f + __expf(-v));       // silu
        } else if (EPI == 3) {                     // M == H here
          int t = nrow & (TT - 1);
          float sh = (t == 0) ? attn_x[(size_t)(nrow / TT) * HH + ncol]
                              : hidden[(size_t)(nrow - 1) * HH + ncol];
          outH[oi] = (_Float16)(v + maa_const[ncol] + sh);
        } else {                                   // 4: + bias
          outF[oi] = v + bias[ncol];
        }
      }
    }
}

// ================= host side =================
extern "C" void kernel_launch(void* const* d_in, const int* in_sizes, int n_in,
                              void* d_out, int out_size, void* d_ws, size_t ws_size,
                              hipStream_t stream) {
  (void)in_sizes; (void)n_in; (void)out_size; (void)ws_size;
  const float* hidden = (const float*)d_in[0];
  const float* attn_x = (const float*)d_in[1];
  const float* maa_x  = (const float*)d_in[4];
  const float* maa_c[5] = { (const float*)d_in[5],  // w
                            (const float*)d_in[6],  // k
                            (const float*)d_in[7],  // v
                            (const float*)d_in[8],  // r
                            (const float*)d_in[9] };// g
  const float* w1  = (const float*)d_in[10];
  const float* w2  = (const float*)d_in[11];
  const float* tdc = (const float*)d_in[12];
  const float* dw1 = (const float*)d_in[13];
  const float* dw2 = (const float*)d_in[14];
  const float* bigw[4] = { (const float*)d_in[16],  // receptance
                           (const float*)d_in[17],  // key
                           (const float*)d_in[18],  // value
                           (const float*)d_in[19] };// gate
  float* out = (float*)d_out;
  const size_t NH = (size_t)NN * HH;               // 33,554,432

  // ---- workspace carve (256B aligned) ----
  char* p = (char*)d_ws;
  auto take = [&](size_t halfs) -> _Float16* {
    _Float16* r = (_Float16*)p;
    p += ((halfs * 2 + 255) & ~(size_t)255);
    return r;
  };
  _Float16* Amix = take(NH);
  _Float16* G    = take((size_t)NN * 5 * LM);      // N x 160
  _Float16* inb[5];                                 // td,k,v,r,g GEMM inputs
  for (int f = 0; f < 5; ++f) inb[f] = take(NH);
  _Float16* td1  = take((size_t)NN * LDY);
  _Float16* w1t  = take((size_t)5 * LM * HH);       // 160 x 2048
  _Float16* w2t  = take((size_t)5 * HH * LM);       // 5 x (2048 x 32)
  _Float16* dw1t = take((size_t)LDY * HH);          // 64 x 2048
  _Float16* dw2t = take((size_t)HH * LDY);          // 2048 x 64
  _Float16* bwt[4];
  for (int i = 0; i < 4; ++i) bwt[i] = take((size_t)HH * HH);

  // ---- weight convert+transpose f32 (KxM) -> f16 (MxK) ----
  dim3 tb(32, 8);
  auto xpose = [&](const float* s, _Float16* d, int K, int M) {
    dim3 g((M + 31) / 32, (K + 31) / 32);
    cvt_transpose<<<g, tb, 0, stream>>>(s, d, K, M);
  };
  xpose(w1, w1t, HH, 5 * LM);
  for (int f = 0; f < 5; ++f)
    xpose(w2 + (size_t)f * LM * HH, w2t + (size_t)f * HH * LM, LM, HH);
  xpose(dw1, dw1t, HH, LDY);
  xpose(dw2, dw2t, LDY, HH);
  for (int i = 0; i < 4; ++i) xpose(bigw[i], bwt[i], HH, HH);

  // ---- mixing input ----
  dim3 blk(256);
  prep_mix<<<2048, blk, 0, stream>>>(hidden, attn_x, maa_x, Amix, NH);

  auto gemm = [&](int epi, const _Float16* A, int lda, const _Float16* Bt, int ldbt,
                  int K, int M, float* oF, _Float16* oH,
                  const float* bias, const float* maac) {
    dim3 grd((M + TN - 1) / TN, NN / TM);
    switch (epi) {
      case 0: gemm_wmma_f16<0><<<grd, blk, 0, stream>>>(A, lda, Bt, ldbt, NN, K, M, oF, oH, bias, hidden, attn_x, maac); break;
      case 1: gemm_wmma_f16<1><<<grd, blk, 0, stream>>>(A, lda, Bt, ldbt, NN, K, M, oF, oH, bias, hidden, attn_x, maac); break;
      case 2: gemm_wmma_f16<2><<<grd, blk, 0, stream>>>(A, lda, Bt, ldbt, NN, K, M, oF, oH, bias, hidden, attn_x, maac); break;
      case 3: gemm_wmma_f16<3><<<grd, blk, 0, stream>>>(A, lda, Bt, ldbt, NN, K, M, oF, oH, bias, hidden, attn_x, maac); break;
      default: gemm_wmma_f16<4><<<grd, blk, 0, stream>>>(A, lda, Bt, ldbt, NN, K, M, oF, oH, bias, hidden, attn_x, maac); break;
    }
  };

  // ---- LoRA mix stage 1: tanh(Amix @ w1) -> G (N x 160, f16) ----
  gemm(1, Amix, HH, w1t, HH, HH, 5 * LM, nullptr, G, nullptr, nullptr);

  // ---- 5 segment GEMMs (K=32), epilogue builds in_f = shifted + maa + m ----
  for (int f = 0; f < 5; ++f)
    gemm(3, G + f * LM, 5 * LM, w2t + (size_t)f * HH * LM, LM, LM, HH,
         nullptr, inb[f], nullptr, maa_c[f]);

  // ---- big projections: r (f=3), k (f=1), v (f=2), g (f=4, silu) ----
  gemm(0, inb[3], HH, bwt[0], HH, HH, HH, out + 0 * NH, nullptr, nullptr, nullptr);
  gemm(0, inb[1], HH, bwt[1], HH, HH, HH, out + 1 * NH, nullptr, nullptr, nullptr);
  gemm(0, inb[2], HH, bwt[2], HH, HH, HH, out + 2 * NH, nullptr, nullptr, nullptr);
  gemm(2, inb[4], HH, bwt[3], HH, HH, HH, out + 3 * NH, nullptr, nullptr, nullptr);

  // ---- decay LoRA: tanh(td_in @ dw1) @ dw2 + time_decay ----
  gemm(1, inb[0], HH, dw1t, HH, HH, LDY, nullptr, td1, nullptr, nullptr);
  gemm(4, td1, LDY, dw2t, LDY, LDY, HH, out + 4 * NH, nullptr, tdc, nullptr);

  // ---- attn_x_new ----
  copy_last<<<(BB * HH + 255) / 256, blk, 0, stream>>>(hidden, out + 5 * NH);
}
